// AttentionBlock_81681688035364
// MI455X (gfx1250) — matrix operand order repaired and checked by reading, not compiled, not run on hardware
//
#include <hip/hip_runtime.h>
#include <hip/hip_bf16.h>

#define BB 64
#define TT 2048
#define DD 512
#define UU 512

typedef __attribute__((ext_vector_type(16))) __bf16 v16bf;
typedef __attribute__((ext_vector_type(8)))  float  v8f;

#if __has_builtin(__builtin_amdgcn_tanhf)
__device__ __forceinline__ float fast_tanh(float x) { return __builtin_amdgcn_tanhf(x); }
#else
__device__ __forceinline__ float fast_tanh(float x) { return tanhf(x); }
#endif

// ------------------------------------------------------------------
// Kernel 1: W1 [D,U] fp32  ->  W1^T [U,D] bf16 (contiguous-K B-fragments)
// ------------------------------------------------------------------
__global__ __launch_bounds__(256) void k_w1t(const float* __restrict__ W1,
                                             __bf16* __restrict__ w1t) {
    int idx = blockIdx.x * 256 + threadIdx.x;   // 0 .. 262143
    int u = idx >> 9;
    int d = idx & 511;
    w1t[idx] = (__bf16)W1[d * UU + u];
}

// ------------------------------------------------------------------
// Kernel 2: ph[b][u] = hidden[b,:] @ W2[:,u] + b2[u]
// ------------------------------------------------------------------
__global__ __launch_bounds__(256) void k_ph(const float* __restrict__ hidden,
                                            const float* __restrict__ W2,
                                            const float* __restrict__ b2,
                                            float* __restrict__ ph) {
    int b = blockIdx.x;
    int u = threadIdx.x;
    float a0 = b2[u];
    float a1 = b2[u + 256];
    const float* h = hidden + b * DD;
    for (int d = 0; d < DD; ++d) {
        float hv = h[d];
        a0 = fmaf(hv, W2[d * UU + u], a0);
        a1 = fmaf(hv, W2[d * UU + u + 256], a1);
    }
    ph[b * UU + u]       = a0;
    ph[b * UU + u + 256] = a1;
}

// ------------------------------------------------------------------
// Kernel 3: scores[b][t] = V . tanh(features[b,t,:] @ W1 + b1 + ph[b]) + bV
// One block = one (b, 32-row T tile). 8 waves; wave w owns N-tiles 4w..4w+3
// and both 16-row M-tiles (B fragment reused across M). bf16 WMMA 16x16x32.
// ------------------------------------------------------------------
__global__ __launch_bounds__(256) void k_score(const float*  __restrict__ features,
                                               const __bf16* __restrict__ w1t,
                                               const float*  __restrict__ b1,
                                               const float*  __restrict__ ph,
                                               const float*  __restrict__ V,
                                               const float*  __restrict__ bV,
                                               float* __restrict__ scores) {
    __shared__ float s_score[32];
    const int tid   = threadIdx.x;
    const int wave  = tid >> 5;
    const int lane  = tid & 31;
    const int lrow  = lane & 15;   // A: M row / B: N col / C: N col
    const int khalf = lane >> 4;   // lane-group selector
    const int b  = blockIdx.y;
    const int t0 = blockIdx.x * 32;

    if (tid < 32) s_score[tid] = 0.0f;
    __syncthreads();

    v8f acc[2][4] = {};

    const float* fbase0 = features + ((size_t)b * TT + t0 + lrow) * DD;
    const float* fbase1 = fbase0 + 16 * DD;

    #pragma unroll 1
    for (int k0 = 0; k0 < DD; k0 += 32) {
        // A fragment (contiguous K, matches B):
        //  lane<16 : elems 0-15 -> K=k0+0..15
        //  lane>=16: elems 0-15 -> K=k0+16..31
        const float* fp0 = fbase0 + k0 + 16 * khalf;
        const float* fp1 = fbase1 + k0 + 16 * khalf;
        v16bf afrag0, afrag1;
        #pragma unroll
        for (int i = 0; i < 8; ++i) {
            afrag0[i]     = (__bf16)fp0[i];
            afrag0[8 + i] = (__bf16)fp0[8 + i];
            afrag1[i]     = (__bf16)fp1[i];
            afrag1[8 + i] = (__bf16)fp1[8 + i];
        }

        #pragma unroll
        for (int j = 0; j < 4; ++j) {
            // B fragment: lane holds col N=lrow, elems i -> K = k0 + 16*khalf + i
            const int n0 = (wave * 4 + j) * 16;
            const __bf16* wp = w1t + (size_t)(n0 + lrow) * DD + k0 + 16 * khalf;
            v16bf bfrag;
            #pragma unroll
            for (int i = 0; i < 16; ++i) bfrag[i] = wp[i];

            acc[0][j] = __builtin_amdgcn_wmma_f32_16x16x32_bf16(
                false, afrag0, false, bfrag, (short)0, acc[0][j], false, false);
            acc[1][j] = __builtin_amdgcn_wmma_f32_16x16x32_bf16(
                false, afrag1, false, bfrag, (short)0, acc[1][j], false, false);
        }
    }

    // Epilogue: tanh(acc + b1 + ph) * V, reduce over N (hardware v_tanh_f32).
    // C/D layout: VGPR r, lane l -> M = r + 8*(l>>4), N = l&15 (+ tile base)
    float partial[2][8];
    #pragma unroll
    for (int m = 0; m < 2; ++m)
        #pragma unroll
        for (int r = 0; r < 8; ++r) partial[m][r] = 0.0f;

    #pragma unroll
    for (int j = 0; j < 4; ++j) {
        int ncol = (wave * 4 + j) * 16 + lrow;
        float bias = b1[ncol] + ph[b * UU + ncol];
        float vv   = V[ncol];
        #pragma unroll
        for (int m = 0; m < 2; ++m) {
            #pragma unroll
            for (int r = 0; r < 8; ++r) {
                float g = fast_tanh(acc[m][j][r] + bias);
                partial[m][r] = fmaf(g, vv, partial[m][r]);
            }
        }
    }

    // Butterfly reduce across the 16 lanes sharing the same M rows.
    #pragma unroll
    for (int msk = 8; msk >= 1; msk >>= 1) {
        #pragma unroll
        for (int m = 0; m < 2; ++m)
            #pragma unroll
            for (int r = 0; r < 8; ++r)
                partial[m][r] += __shfl_xor(partial[m][r], msk, 32);
    }

    if (lrow == 0) {
        #pragma unroll
        for (int m = 0; m < 2; ++m)
            #pragma unroll
            for (int r = 0; r < 8; ++r)
                atomicAdd(&s_score[m * 16 + 8 * khalf + r], partial[m][r]);  // ds_add_f32
    }
    __syncthreads();

    if (tid < 32)
        scores[(size_t)b * TT + t0 + tid] = s_score[tid] + bV[0];
}

// ------------------------------------------------------------------
// Kernel 4: softmax over T per batch; writes attention weights to d_out.
// ------------------------------------------------------------------
__global__ __launch_bounds__(256) void k_softmax(const float* __restrict__ scores,
                                                 float* __restrict__ weights) {
    __shared__ float red[256];
    const int b   = blockIdx.x;
    const int tid = threadIdx.x;
    const float* s = scores + (size_t)b * TT;

    float local[8];
    float mx = -3.402823466e38f;
    #pragma unroll
    for (int i = 0; i < 8; ++i) {
        local[i] = s[tid + i * 256];
        mx = fmaxf(mx, local[i]);
    }
    red[tid] = mx;
    __syncthreads();
    for (int off = 128; off >= 1; off >>= 1) {
        if (tid < off) red[tid] = fmaxf(red[tid], red[tid + off]);
        __syncthreads();
    }
    mx = red[0];
    __syncthreads();

    float sum = 0.0f;
    #pragma unroll
    for (int i = 0; i < 8; ++i) {
        local[i] = __expf(local[i] - mx);
        sum += local[i];
    }
    red[tid] = sum;
    __syncthreads();
    for (int off = 128; off >= 1; off >>= 1) {
        if (tid < off) red[tid] += red[tid + off];
        __syncthreads();
    }
    float inv = 1.0f / red[0];

    float* w = weights + (size_t)b * TT;
    #pragma unroll
    for (int i = 0; i < 8; ++i) w[tid + i * 256] = local[i] * inv;
}

// ------------------------------------------------------------------
// Kernel 5: context[b][d] = sum_t w[b][t] * features[b][t][d]  (HBM-bound)
// ------------------------------------------------------------------
__global__ __launch_bounds__(256) void k_context(const float* __restrict__ features,
                                                 const float* __restrict__ weights,
                                                 float* __restrict__ ctx) {
    __shared__ float wsh[256];
    const int b = blockIdx.y;
    const int d = blockIdx.x * 256 + threadIdx.x;
    const float* w = weights + (size_t)b * TT;

    float acc = 0.0f;
    for (int tc = 0; tc < TT; tc += 256) {
        __syncthreads();
        wsh[threadIdx.x] = w[tc + threadIdx.x];
        __syncthreads();
        const float* f = features + ((size_t)b * TT + tc) * DD + d;
        #pragma unroll 8
        for (int i = 0; i < 256; ++i)
            acc = fmaf(wsh[i], f[(size_t)i * DD], acc);
    }
    ctx[b * DD + d] = acc;
}

// ------------------------------------------------------------------
extern "C" void kernel_launch(void* const* d_in, const int* in_sizes, int n_in,
                              void* d_out, int out_size, void* d_ws, size_t ws_size,
                              hipStream_t stream) {
    const float* features = (const float*)d_in[0];
    const float* hidden   = (const float*)d_in[1];
    const float* W1       = (const float*)d_in[2];
    const float* b1       = (const float*)d_in[3];
    const float* W2       = (const float*)d_in[4];
    const float* b2       = (const float*)d_in[5];
    const float* V        = (const float*)d_in[6];
    const float* bV       = (const float*)d_in[7];

    char* ws = (char*)d_ws;
    __bf16* w1t   = (__bf16*)ws;                         // 512 KiB: bf16 W1^T [U][D]
    float*  ph    = (float*)(ws + (512u << 10));         // 128 KiB: [B][U]
    float*  score = (float*)(ws + (640u << 10));         // 512 KiB: [B][T]

    float* ctx     = (float*)d_out;            // [B, D]
    float* weights = (float*)d_out + BB * DD;  // [B, T, 1]

    k_w1t    <<<dim3((DD * UU) / 256), 256, 0, stream>>>(W1, w1t);
    k_ph     <<<dim3(BB),              256, 0, stream>>>(hidden, W2, b2, ph);
    k_score  <<<dim3(TT / 32, BB),     256, 0, stream>>>(features, w1t, b1, ph, V, bV, score);
    k_softmax<<<dim3(BB),              256, 0, stream>>>(score, weights);
    k_context<<<dim3(DD / 256, BB),    256, 0, stream>>>(features, weights, ctx);
}
